// MACBlock_51110110822849
// MI455X (gfx1250) — matrix operand order repaired and checked by reading, not compiled, not run on hardware
//
#include <hip/hip_runtime.h>
#include <hip/hip_bf16.h>

// ---------------------------------------------------------------------------
// MI455X (gfx1250, wave32) implementation of the MAC block.
// All dense math runs through v_wmma_f32_16x16x32_f16 (fp32 accum, f16 mul).
// GEMM uses double-buffered LDS staging + global_prefetch_b8 warmup.
// ---------------------------------------------------------------------------

typedef __attribute__((ext_vector_type(16))) _Float16 v16h;
typedef __attribute__((ext_vector_type(8)))  float    v8f;
typedef __attribute__((ext_vector_type(2)))  __fp16   fp16x2;

union HF { unsigned int u[8]; v16h h; };

__device__ __forceinline__ unsigned int pack2(float a, float b) {
#if __has_builtin(__builtin_amdgcn_cvt_pkrtz)
  union { fp16x2 v; unsigned int u; } x;
  x.v = __builtin_amdgcn_cvt_pkrtz(a, b);   // v_cvt_pk_rtz_f16_f32
  return x.u;
#else
  union { _Float16 h[2]; unsigned int u; } x;
  x.h[0] = (_Float16)a; x.h[1] = (_Float16)b;
  return x.u;
#endif
}

// ---------------- generic strided GEMM via WMMA ----------------
// C[m,n] = act( sum_k A[m*arm + k*ark] * B[k*brk + n*brn] + bias[n] )
// Tiles: 128x128x32, 256 threads (8 waves), each wave -> 32x64 patch.
// Double-buffered LDS; next tile fetched to registers during WMMA burst.
__global__ __launch_bounds__(256) void gemm_wmma(
    const float* __restrict__ A, long arm, long ark,
    const float* __restrict__ B, long brk, long brn,
    const float* __restrict__ bias, float* __restrict__ C,
    int M, int N, int K, int act)
{
  __shared__ __align__(16) unsigned int As[2][128][16]; // [buf][m][k-pair]
  __shared__ __align__(16) unsigned int Bs[2][16][128]; // [buf][k-pair][n]
  const int tid  = threadIdx.x;
  const int lane = tid & 31;
  const int wave = tid >> 5;
  const int wm   = (wave >> 1) * 32;   // 4 waves along M
  const int wn   = (wave & 1) * 64;    // 2 waves along N
  const int bm   = blockIdx.y * 128;
  const int bn   = blockIdx.x * 128;
  const int half = lane >> 4;
  const int l16  = lane & 15;

  // per-thread staging coordinates (constant across the 8 steps)
  const int ap  = tid & 15;        // A k-pair index
  const int am0 = tid >> 4;        // A row base; rows am0 + 16*t
  const int bn0 = tid & 127;       // B column
  const int bp0 = tid >> 7;        // B k-pair base; pairs bp0 + 2*t

  float aLo[8], aHi[8], bLo[8], bHi[8];

  auto fetch = [&](int k0) {
#pragma unroll
    for (int t = 0; t < 8; ++t) {
      int gm = bm + am0 + 16 * t; if (gm >= M) gm = M - 1;
      long base = (long)gm * arm + (long)(k0 + 2 * ap) * ark;
      aLo[t] = A[base]; aHi[t] = A[base + ark];
    }
#pragma unroll
    for (int t = 0; t < 8; ++t) {
      long base = (long)(k0 + 2 * (bp0 + 2 * t)) * brk + (long)(bn + bn0) * brn;
      bLo[t] = B[base]; bHi[t] = B[base + brk];
    }
  };
  auto stage = [&](int buf) {
#pragma unroll
    for (int t = 0; t < 8; ++t) As[buf][am0 + 16 * t][ap] = pack2(aLo[t], aHi[t]);
#pragma unroll
    for (int t = 0; t < 8; ++t) Bs[buf][bp0 + 2 * t][bn0] = pack2(bLo[t], bHi[t]);
  };

  v8f acc[2][4];
  for (int a0 = 0; a0 < 2; ++a0)
    for (int b0 = 0; b0 < 4; ++b0)
      for (int e = 0; e < 8; ++e) acc[a0][b0][e] = 0.f;

  fetch(0);
  stage(0);

  int buf = 0;
  for (int k0 = 0; k0 < K; k0 += 32, buf ^= 1) {
    __syncthreads();
    const bool next = (k0 + 32) < K;
    if (next) {
      // L2 warm-up two tiles ahead (global_prefetch_b8)
      if (k0 + 64 < K) {
        int gm = bm + am0; if (gm >= M) gm = M - 1;
        __builtin_prefetch(&A[(long)gm * arm + (long)(k0 + 64) * ark], 0, 1);
        __builtin_prefetch(&B[(long)(k0 + 64 + 2 * bp0) * brk + (long)(bn + bn0) * brn], 0, 1);
      }
      fetch(k0 + 32);                 // global loads issue now, waited after WMMAs
    }

    // A fragments: ISA 16-bit A 16x32 layout -> two b128 LDS reads.
    HF af[2];
#pragma unroll
    for (int mi = 0; mi < 2; ++mi) {
      const uint4* pr = (const uint4*)As[buf][wm + mi * 16 + l16];
      uint4 x0 = pr[half];          // K 0..7  (lanes<16) / 8..15
      uint4 x1 = pr[2 + half];      // K 16..23 / 24..31
      af[mi].u[0] = x0.x; af[mi].u[1] = x0.y; af[mi].u[2] = x0.z; af[mi].u[3] = x0.w;
      af[mi].u[4] = x1.x; af[mi].u[5] = x1.y; af[mi].u[6] = x1.z; af[mi].u[7] = x1.w;
    }
    // B fragments: lanes 0-15 hold K 0..7,16..23; lanes 16-31 K 8..15,24..31.
    HF bf[4];
#pragma unroll
    for (int ni = 0; ni < 4; ++ni) {
      int col = wn + ni * 16 + l16;
      int p0 = half * 4;
#pragma unroll
      for (int i = 0; i < 8; ++i) bf[ni].u[i] = Bs[buf][p0 + (i & 3) + (i >> 2) * 8][col];
    }
#pragma unroll
    for (int mi = 0; mi < 2; ++mi)
#pragma unroll
      for (int ni = 0; ni < 4; ++ni)
        acc[mi][ni] = __builtin_amdgcn_wmma_f32_16x16x32_f16(
            false, af[mi].h, false, bf[ni].h, (short)0, acc[mi][ni], false, false);

    if (next) stage(buf ^ 1);         // convert + write next tile to other buffer
  }

  // epilogue: C layout -> lane = n, vgpr i = m (+8 for upper lane half)
  for (int ni = 0; ni < 4; ++ni) {
    int n = bn + wn + ni * 16 + l16;
    float bv = bias ? bias[n] : 0.f;
    for (int mi = 0; mi < 2; ++mi) {
      for (int i = 0; i < 8; ++i) {
        int m = bm + wm + mi * 16 + i + half * 8;
        if (m < M) {
          float v = acc[mi][ni][i] + bv;
          if (act == 1) {                       // SiLU
            v = v / (1.f + __expf(-v));
          } else if (act == 2) {                // tanh GELU
            float c = 0.7978845608f * (v + 0.044715f * v * v * v);
            v = 0.5f * v * (1.f + tanhf(c));
          }
          C[(long)m * N + n] = v;
        }
      }
    }
  }
}

// ---------------- flash attention (causal, 2064 tokens) ----------------
#define NTOK   2064
#define QSTART 1040

__global__ __launch_bounds__(128) void flash_attn(
    const float* __restrict__ aqkv, float* __restrict__ o)
{
  __shared__ __align__(16) _Float16 Pst[4][16][32];  // per-wave P staging
  const int lane = threadIdx.x & 31;
  const int w    = threadIdx.x >> 5;
  const int gt   = blockIdx.x * 4 + w;   // 0..2047
  const int qt   = gt & 63;              // 16-row query tile within batch
  const int h    = (gt >> 6) & 15;
  const int b    = gt >> 10;
  const int half = lane >> 4;
  const int l16  = lane & 15;
  const float scale = 0.125f;            // 1/sqrt(64)

  const float* base = aqkv + (long)b * NTOK * 3072 + h * 64;

  // Q fragment (A layout, pre-scaled), two d-halves of dh=64
  HF qf[2];
  {
    const float* qrow = base + (long)(QSTART + qt * 16 + l16) * 3072;
#pragma unroll
    for (int hd = 0; hd < 2; ++hd) {
      int d0 = hd * 32 + half * 16;
#pragma unroll
      for (int i = 0; i < 4; ++i)
        qf[hd].u[i] = pack2(qrow[d0 + 2 * i] * scale, qrow[d0 + 2 * i + 1] * scale);
#pragma unroll
      for (int i = 0; i < 4; ++i)
        qf[hd].u[4 + i] = pack2(qrow[d0 + 8 + 2 * i] * scale, qrow[d0 + 8 + 2 * i + 1] * scale);
    }
  }

  v8f ov[4];
  for (int sl = 0; sl < 4; ++sl) for (int e = 0; e < 8; ++e) ov[sl][e] = 0.f;
  float rm[8], rl[8];
  for (int i = 0; i < 8; ++i) { rm[i] = -3.0e38f; rl[i] = 0.f; }

  const int qlast = QSTART + qt * 16 + 15;
  for (int kb = 0; kb <= qlast; kb += 32) {
    // ---- S = (Q*scale) K^T over two 16-key halves ----
    v8f S[2];
#pragma unroll
    for (int nh = 0; nh < 2; ++nh) {
      int keyi = kb + nh * 16 + l16;
      int keyc = keyi > NTOK - 1 ? NTOK - 1 : keyi;
      const float* krow = base + 1024 + (long)keyc * 3072;
      HF kf0, kf1;
      int d0 = half * 16;
#pragma unroll
      for (int i = 0; i < 8; ++i) kf0.u[i] = pack2(krow[d0 + 2 * i], krow[d0 + 2 * i + 1]);
#pragma unroll
      for (int i = 0; i < 8; ++i) kf1.u[i] = pack2(krow[32 + d0 + 2 * i], krow[32 + d0 + 2 * i + 1]);
      v8f s;
      for (int e = 0; e < 8; ++e) s[e] = 0.f;
      s = __builtin_amdgcn_wmma_f32_16x16x32_f16(false, qf[0].h, false, kf0.h, (short)0, s, false, false);
      s = __builtin_amdgcn_wmma_f32_16x16x32_f16(false, qf[1].h, false, kf1.h, (short)0, s, false, false);
#pragma unroll
      for (int i = 0; i < 8; ++i) {
        int qtok = QSTART + qt * 16 + i + half * 8;
        if (keyi > qtok) s[i] = -1.0e9f;     // causal mask
      }
      S[nh] = s;
    }

    // ---- online softmax update (rows live in vgpr index, keys in lanes) ----
    float pr0[8], pr1[8];
#pragma unroll
    for (int i = 0; i < 8; ++i) {
      float a = fmaxf(S[0][i], S[1][i]);
      for (int off = 8; off >= 1; off >>= 1) a = fmaxf(a, __shfl_xor(a, off, 32));
      float nm  = fmaxf(rm[i], a);
      float scl = __expf(rm[i] - nm);
      rm[i] = nm;
      float p0 = __expf(S[0][i] - nm);
      float p1 = __expf(S[1][i] - nm);
      pr0[i] = p0; pr1[i] = p1;
      float rs = p0 + p1;
      for (int off = 8; off >= 1; off >>= 1) rs += __shfl_xor(rs, off, 32);
      rl[i] = rl[i] * scl + rs;
      for (int sl = 0; sl < 4; ++sl) ov[sl][i] *= scl;
    }

    // ---- stage P through LDS: C-layout -> A-layout (16x32 f16) ----
#pragma unroll
    for (int i = 0; i < 8; ++i) {
      int m = i + half * 8;
      Pst[w][m][l16]      = (_Float16)pr0[i];
      Pst[w][m][16 + l16] = (_Float16)pr1[i];
    }
    asm volatile("s_wait_dscnt 0" ::: "memory");
    HF pf;
    {
      const unsigned int* pr = (const unsigned int*)Pst[w][l16];
      int jb = half * 8;
#pragma unroll
      for (int i = 0; i < 4; ++i) pf.u[i]     = pr[jb + i];
#pragma unroll
      for (int i = 0; i < 4; ++i) pf.u[4 + i] = pr[4 + jb + i];
    }

    // ---- O += P @ V over four 16-wide d slices ----
#pragma unroll
    for (int sl = 0; sl < 4; ++sl) {
      HF vf;
      int d = sl * 16 + l16;
      int koff = half * 16;
#pragma unroll
      for (int i = 0; i < 8; ++i) {
        int k0i = kb + koff + 2 * i;
        int k1i = k0i + 1;
        if (k0i > NTOK - 1) k0i = NTOK - 1;
        if (k1i > NTOK - 1) k1i = NTOK - 1;
        vf.u[i] = pack2(base[2048 + (long)k0i * 3072 + d],
                        base[2048 + (long)k1i * 3072 + d]);
      }
      ov[sl] = __builtin_amdgcn_wmma_f32_16x16x32_f16(
          false, pf.h, false, vf.h, (short)0, ov[sl], false, false);
    }
  }

  // ---- write O / l, heads re-merged: o[b, n, h*64 + d] ----
#pragma unroll
  for (int i = 0; i < 8; ++i) {
    float inv = 1.f / rl[i];
    int m = qt * 16 + i + half * 8;                     // 0..1023 within batch
    long orow = ((long)(b * 1024 + m)) * 1024 + h * 64;
    for (int sl = 0; sl < 4; ++sl)
      o[orow + sl * 16 + l16] = ov[sl][i] * inv;
  }
}

// ---------------- elementwise helpers ----------------
__global__ void k_silu(const float* __restrict__ pre, float* __restrict__ hh, int n) {
  int i = blockIdx.x * blockDim.x + threadIdx.x;
  if (i < n) { float v = pre[i]; hh[i] = v / (1.f + __expf(-v)); }
}
__global__ void k_dloss(const float* __restrict__ y, const float* __restrict__ v,
                        float* __restrict__ d, float s, int n) {
  int i = blockIdx.x * blockDim.x + threadIdx.x;
  if (i < n) d[i] = s * (y[i] - v[i]);
}
__global__ void k_dsilu(float* __restrict__ dh, const float* __restrict__ pre, int n) {
  int i = blockIdx.x * blockDim.x + threadIdx.x;
  if (i < n) {
    float x = pre[i];
    float sg = 1.f / (1.f + __expf(-x));
    dh[i] *= sg * (1.f + x * (1.f - sg));
  }
}
__global__ void k_colsum(const float* __restrict__ X, float* __restrict__ o,
                         int rows, int cols) {
  int n = blockIdx.x * blockDim.x + threadIdx.x;
  if (n >= cols) return;
  float s = 0.f;
  for (int r = 0; r < rows; ++r) s += X[(long)r * cols + n];
  o[n] = s;
}
__global__ void k_update(float* __restrict__ p, float* __restrict__ m,
                         const float* __restrict__ g, int n) {
  int i = blockIdx.x * blockDim.x + threadIdx.x;
  if (i < n) {
    float mm = 0.9f * m[i] - 0.1f * g[i];
    m[i] = mm;
    p[i] = 0.98f * p[i] + mm;
  }
}
__global__ void k_pack_chunk(const float* __restrict__ qkv, float* __restrict__ kc,
                             float* __restrict__ vc, int c) {
  int idx = blockIdx.x * blockDim.x + threadIdx.x;   // 256*1024
  if (idx >= 256 * 1024) return;
  int r = idx >> 10, d = idx & 1023;
  int bb = r >> 7, t = r & 127;
  long src = ((long)(bb * 1024 + c * 128 + t)) * 3072;
  kc[idx] = qkv[src + 1024 + d];
  vc[idx] = qkv[src + 2048 + d];
}
__global__ void k_tokens(const float* __restrict__ P, const float* __restrict__ mt,
                         const float* __restrict__ x, float* __restrict__ tok) {
  int idx = blockIdx.x * blockDim.x + threadIdx.x;   // 4128*1024
  if (idx >= 4128 * 1024) return;
  int row = idx >> 10, d = idx & 1023;
  int bb = row / NTOK, n = row % NTOK;
  float v;
  if (n < 16)        v = P[n * 1024 + d];
  else if (n < 1040) v = mt[((long)(bb * 1024 + n - 16)) * 1024 + d];
  else               v = x[((long)(bb * 1024 + n - 1040)) * 1024 + d];
  tok[idx] = v;
}
__global__ void k_mul(const float* __restrict__ a, const float* __restrict__ b,
                      float* __restrict__ o, int n) {
  int i = blockIdx.x * blockDim.x + threadIdx.x;
  if (i < n) o[i] = a[i] * b[i];
}

// ---------------------------------------------------------------------------
extern "C" void kernel_launch(void* const* d_in, const int* in_sizes, int n_in,
                              void* d_out, int out_size, void* d_ws, size_t ws_size,
                              hipStream_t stream) {
  const float* x    = (const float*)d_in[0];
  const float* Wqkv = (const float*)d_in[1];
  const float* mW1  = (const float*)d_in[2];
  const float* mb1  = (const float*)d_in[3];
  const float* mW2  = (const float*)d_in[4];
  const float* mb2  = (const float*)d_in[5];
  const float* Pp   = (const float*)d_in[6];
  const float* Wqa  = (const float*)d_in[7];
  const float* Wo   = (const float*)d_in[8];
  const float* ffW1 = (const float*)d_in[9];
  const float* ffb1 = (const float*)d_in[10];
  const float* ffW2 = (const float*)d_in[11];
  const float* ffb2 = (const float*)d_in[12];
  float* out = (float*)d_out;
  float* ws  = (float*)d_ws;

  // -------- workspace layout (floats) --------
  float* qkv    = ws;                    // 2048x3072
  float* kc     = qkv    + 6291456;      // 256x1024
  float* vc     = kc     + 262144;       // 256x1024
  float* pre1   = vc     + 262144;       // 256x512
  float* hbuf   = pre1   + 131072;       // 256x512
  float* ybuf   = hbuf   + 131072;       // 256x1024
  float* dbuf   = ybuf   + 262144;       // 256x1024
  float* dhb    = dbuf   + 262144;       // 256x512
  float* W1p    = dhb    + 131072;       // 512x1024
  float* b1p    = W1p    + 524288;       // 512
  float* W2p    = b1p    + 512;          // 1024x512
  float* b2p    = W2p    + 524288;       // 1024
  float* W1m    = b2p    + 1024;         // momenta (contiguous block)
  float* b1m    = W1m    + 524288;
  float* W2m    = b1m    + 512;
  float* b2m    = W2m    + 524288;
  float* W1g    = b2m    + 1024;         // grads
  float* b1g    = W1g    + 524288;
  float* W2g    = b1g    + 512;
  float* b2g    = W2g    + 524288;
  float* hq     = b2g    + 1024;         // 2048x512
  float* memtok = hq     + 1048576;      // 2048x1024
  float* tokens = memtok + 2097152;      // 4128x1024
  float* aqkv   = tokens + 4227072;      // 4128x3072
  float* obuf   = aqkv   + 12681216;     // 2048x1024
  float* y2     = obuf   + 2097152;      // 2048x1024
  // aliases over dead ranges:
  float* y1     = tokens;                // 2048x1024 (tokens dead after aqkv)
  float* ff1    = aqkv;                  // 2048x4096 (aqkv dead after flash)
  float* newQ   = memtok;                // 2048x1024 (memtok dead after tokens)
  float* hq2    = hq;                    // 2048x512
  float* memout = obuf;                  // 2048x1024 (obuf dead after y1)

  auto gemm = [&](const float* A, long arm, long ark,
                  const float* B, long brk, long brn,
                  const float* bias, float* C, int M, int N, int K, int act) {
    dim3 g((N + 127) / 128, (M + 127) / 128);
    gemm_wmma<<<g, 256, 0, stream>>>(A, arm, ark, B, brk, brn, bias, C, M, N, K, act);
  };
  auto ew = [&](int n) { return dim3((n + 255) / 256); };

  // init memory-MLP params + zero momenta (deterministic each call)
  (void)hipMemcpyAsync(W1p, mW1, 524288 * 4, hipMemcpyDeviceToDevice, stream);
  (void)hipMemcpyAsync(b1p, mb1, 512 * 4,    hipMemcpyDeviceToDevice, stream);
  (void)hipMemcpyAsync(W2p, mW2, 524288 * 4, hipMemcpyDeviceToDevice, stream);
  (void)hipMemcpyAsync(b2p, mb2, 1024 * 4,   hipMemcpyDeviceToDevice, stream);
  (void)hipMemsetAsync(W1m, 0, (524288 + 512 + 524288 + 1024) * 4, stream);

  // qkv = x @ Wqkv.T   (2048x3072)
  gemm(x, 1024, 1, Wqkv, 1, 1024, nullptr, qkv, 2048, 3072, 1024, 0);

  // -------- inner memory-update scan: 8 chunks --------
  const float dscale = 2.f / 262144.f;
  for (int c = 0; c < 8; ++c) {
    k_pack_chunk<<<ew(262144), 256, 0, stream>>>(qkv, kc, vc, c);
    // fwd: pre1 = k @ W1.T + b1 ; h = silu(pre1) ; y = h @ W2.T + b2
    gemm(kc, 1024, 1, W1p, 1, 1024, b1p, pre1, 256, 512, 1024, 0);
    k_silu<<<ew(131072), 256, 0, stream>>>(pre1, hbuf, 131072);
    gemm(hbuf, 512, 1, W2p, 1, 512, b2p, ybuf, 256, 1024, 512, 0);
    // d = 2*(y - v)/numel
    k_dloss<<<ew(262144), 256, 0, stream>>>(ybuf, vc, dbuf, dscale, 262144);
    // dW2 = d.T @ h ; db2 = colsum(d)
    gemm(dbuf, 1, 1024, hbuf, 512, 1, nullptr, W2g, 1024, 512, 256, 0);
    k_colsum<<<ew(1024), 256, 0, stream>>>(dbuf, b2g, 256, 1024);
    // dh = (d @ W2) * silu'(pre1)
    gemm(dbuf, 1024, 1, W2p, 512, 1, nullptr, dhb, 256, 512, 1024, 0);
    k_dsilu<<<ew(131072), 256, 0, stream>>>(dhb, pre1, 131072);
    // dW1 = dh.T @ k ; db1 = colsum(dh)
    gemm(dhb, 1, 512, kc, 1024, 1, nullptr, W1g, 512, 1024, 256, 0);
    k_colsum<<<ew(512), 256, 0, stream>>>(dhb, b1g, 256, 512);
    // momentum + decayed param update
    k_update<<<ew(524288), 256, 0, stream>>>(W1p, W1m, W1g, 524288);
    k_update<<<ew(512),    256, 0, stream>>>(b1p, b1m, b1g, 512);
    k_update<<<ew(524288), 256, 0, stream>>>(W2p, W2m, W2g, 524288);
    k_update<<<ew(1024),   256, 0, stream>>>(b2p, b2m, b2g, 1024);
  }

  // -------- mem_tok = mem_read(Q, pf) --------
  gemm(qkv, 3072, 1, W1p, 1, 1024, b1p, hq, 2048, 512, 1024, 1);   // fused SiLU
  gemm(hq, 512, 1, W2p, 1, 512, b2p, memtok, 2048, 1024, 512, 0);

  // -------- tokens, attention qkv, flash attention --------
  k_tokens<<<ew(4128 * 1024), 256, 0, stream>>>(Pp, memtok, x, tokens);
  gemm(tokens, 1024, 1, Wqa, 1, 1024, nullptr, aqkv, 4128, 3072, 1024, 0);
  flash_attn<<<512, 128, 0, stream>>>(aqkv, obuf);

  // -------- output projection + FFN (only the 2048 needed rows) --------
  gemm(obuf, 1024, 1, Wo, 1, 1024, nullptr, y1, 2048, 1024, 1024, 0);
  gemm(y1, 1024, 1, ffW1, 1, 1024, ffb1, ff1, 2048, 4096, 1024, 2); // fused GELU
  gemm(ff1, 4096, 1, ffW2, 1, 4096, ffb2, y2, 2048, 1024, 4096, 0);

  // -------- final gate: out = y2 * mem_read(y2 @ Wqkv.T[:D], pf) --------
  gemm(y2, 1024, 1, Wqkv, 1, 1024, nullptr, newQ, 2048, 1024, 1024, 0);
  gemm(newQ, 1024, 1, W1p, 1, 1024, b1p, hq2, 2048, 512, 1024, 1);  // fused SiLU
  gemm(hq2, 512, 1, W2p, 1, 512, b2p, memout, 2048, 1024, 512, 0);
  k_mul<<<ew(2097152), 256, 0, stream>>>(y2, memout, out, 2097152);
}
